// DeepFM_26027501814310
// MI455X (gfx1250) — compile-verified, matrix operand order checked by reading, not run on hardware
//
#include <hip/hip_runtime.h>
#include <hip/hip_bf16.h>

// ---------------------------------------------------------------- types
typedef __attribute__((ext_vector_type(16))) _Float16 v16h;
typedef __attribute__((ext_vector_type(8)))  _Float16 v8h;
typedef __attribute__((ext_vector_type(8)))  float    v8f;

union Frag16 { v16h v; v8h h[2]; };

__device__ __forceinline__ void load_frag(Frag16& f, const _Float16* p) {
    f.h[0] = *(const v8h*)(p);
    f.h[1] = *(const v8h*)(p + 16);
}

// ---------------------------------------------------------------- sizes
constexpr int  BB      = 16384;
constexpr int  N_NUM   = 13;
constexpr int  N_FIELDS= 26;
constexpr int  VOCAB   = 100000;
constexpr int  KDIM    = 64;
constexpr int  H1      = 1024;
constexpr int  H2      = 512;
constexpr int  H3      = 256;
constexpr int  DEEP_IN = N_NUM + N_FIELDS * KDIM;   // 1677
constexpr int  KP1     = 1696;                       // pad to multiple of 32

// workspace layout (bytes, 256-aligned)
constexpr size_t OFF_DEEP = 0;                                   // f16 [B,KP1]   (reused as h2)
constexpr size_t OFF_HA   = OFF_DEEP + (size_t)BB * KP1 * 2;     // f16 [B,H1]    (reused as h3)
constexpr size_t OFF_W1T  = OFF_HA   + (size_t)BB * H1  * 2;     // f16 [H1,KP1]
constexpr size_t OFF_W2T  = OFF_W1T  + (size_t)H1 * KP1 * 2;     // f16 [H2,H1]
constexpr size_t OFF_W3T  = OFF_W2T  + (size_t)H2 * H1  * 2;     // f16 [H3,H2]
constexpr size_t OFF_PART = OFF_W3T  + (size_t)H3 * H2  * 2;     // f32 [B]

// ------------------------------------------------ weight transpose+cvt
// W[Kdim,N] fp32 row-major  ->  Wt[N,Kpad] f16, zero-padded in K.
__global__ __launch_bounds__(256) void transpose_cvt(
    const float* __restrict__ W, _Float16* __restrict__ Wt,
    int Kdim, int N, int Kpad)
{
    size_t gid   = (size_t)blockIdx.x * blockDim.x + threadIdx.x;
    size_t total = (size_t)N * Kpad;
    if (gid >= total) return;
    int n  = (int)(gid / Kpad);
    int kk = (int)(gid % Kpad);
    float v = (kk < Kdim) ? W[(size_t)kk * N + n] : 0.0f;
    Wt[(size_t)n * Kpad + kk] = (_Float16)v;
}

// ------------------------------------------------ feature build + FM scalar
// One 64-thread block per batch row; thread k = embedding column k.
__global__ __launch_bounds__(64) void build_features(
    const float* __restrict__ x_num, const int* __restrict__ x_cat,
    const float* __restrict__ bias,  const float* __restrict__ w_num,
    const float* __restrict__ lin_cat, const float* __restrict__ v_num,
    const float* __restrict__ v_cat,
    _Float16* __restrict__ deep, float* __restrict__ partial)
{
    const int b = blockIdx.x;
    const int k = threadIdx.x;                  // 0..63
    __shared__ float red[KDIM];
    __shared__ int   idx[N_FIELDS];
    __shared__ float xn[N_NUM];

    if (k < N_FIELDS) idx[k] = x_cat[b * N_FIELDS + k];
    if (k < N_NUM)    xn[k]  = x_num[b * N_NUM + k];
    __syncthreads();

    float sum = 0.0f, sq = 0.0f;
#pragma unroll
    for (int i = 0; i < N_NUM; ++i) {
        float v = xn[i] * v_num[i * KDIM + k];
        sum += v; sq += v * v;
    }
    _Float16* drow = deep + (size_t)b * KP1;
#pragma unroll 1
    for (int f = 0; f < N_FIELDS; ++f) {
        const float* e = v_cat + ((size_t)f * VOCAB + idx[f]) * KDIM;
        float v = e[k];                          // coalesced across k
        sum += v; sq += v * v;
        drow[N_NUM + f * KDIM + k] = (_Float16)v;
    }
    red[k] = 0.5f * (sum * sum - sq);

    if (k < N_NUM)       drow[k] = (_Float16)xn[k];               // cols 0..12
    else if (k < 32)     drow[DEEP_IN - 13 + k] = (_Float16)0.0f; // pad 1677..1695
    __syncthreads();

    if (k == 0) {
        float inter = 0.0f;
        for (int i = 0; i < KDIM; ++i) inter += red[i];
        float lin = bias[0];
        for (int i = 0; i < N_NUM; ++i)    lin += xn[i] * w_num[i];
        for (int f = 0; f < N_FIELDS; ++f) lin += lin_cat[(size_t)f * VOCAB + idx[f]];
        partial[b] = lin + inter;
    }
}

// ------------------------------------------------ WMMA GEMM + bias + ReLU
// C[M,N] f16 = relu(A[M,Kp] f16 @ B + bias), B supplied transposed Bt[N,Kp].
// Block = 8 waves, 4(M) x 2(N); wave tile = 32M x 64N = 8 wmma accumulators.
// Register double-buffered K pipeline: next K-tile fragments are loaded
// before the current tile's WMMAs so loads overlap matrix math.
__global__ __launch_bounds__(256) void gemm_relu_f16(
    const _Float16* __restrict__ A,  const _Float16* __restrict__ Bt,
    const float* __restrict__ bias,  _Float16* __restrict__ C,
    int M, int N, int Kp, int do_relu)
{
    const int lid = threadIdx.x & 31;
    const int wv  = threadIdx.x >> 5;
    const int wm  = wv & 3;                      // 4 waves in M
    const int wn  = wv >> 2;                     // 2 waves in N
    const int m0  = blockIdx.y * 128 + wm * 32;
    const int n0  = blockIdx.x * 128 + wn * 64;

    // 16-bit A/B 16x32 fragment: lanes 0-15 hold K 0..7 & 16..23 of row (lane&15),
    // lanes 16-31 hold K 8..15 & 24..31. Two b128 loads per lane per fragment.
    const int half16 = lid >> 4;                 // 0 | 1
    const int koff   = half16 << 3;              // 0 | 8
    const _Float16* aptr0 = A  + (size_t)(m0 + (lid & 15)) * Kp + koff;
    const _Float16* aptr1 = aptr0 + (size_t)16 * Kp;
    const _Float16* bptr  = Bt + (size_t)(n0 + (lid & 15)) * Kp + koff;

    v8f acc[2][4] = {};

    Frag16 a0, a1, b[4];
    load_frag(a0, aptr0);
    load_frag(a1, aptr1);
#pragma unroll
    for (int t = 0; t < 4; ++t) load_frag(b[t], bptr + (size_t)(t * 16) * Kp);

    for (int k = 32; k < Kp; k += 32) {
        Frag16 na0, na1, nb[4];
        load_frag(na0, aptr0 + k);
        load_frag(na1, aptr1 + k);
#pragma unroll
        for (int t = 0; t < 4; ++t) load_frag(nb[t], bptr + (size_t)(t * 16) * Kp + k);
        __builtin_prefetch(aptr0 + k + 32, 0, 1);          // global_prefetch_b8
        __builtin_prefetch(bptr  + k + 32, 0, 1);
#pragma unroll
        for (int t = 0; t < 4; ++t) {
            acc[0][t] = __builtin_amdgcn_wmma_f32_16x16x32_f16(
                false, a0.v, false, b[t].v, (short)0, acc[0][t], false, false);
            acc[1][t] = __builtin_amdgcn_wmma_f32_16x16x32_f16(
                false, a1.v, false, b[t].v, (short)0, acc[1][t], false, false);
        }
        a0 = na0; a1 = na1;
#pragma unroll
        for (int t = 0; t < 4; ++t) b[t] = nb[t];
    }
    // epilogue K-tile
#pragma unroll
    for (int t = 0; t < 4; ++t) {
        acc[0][t] = __builtin_amdgcn_wmma_f32_16x16x32_f16(
            false, a0.v, false, b[t].v, (short)0, acc[0][t], false, false);
        acc[1][t] = __builtin_amdgcn_wmma_f32_16x16x32_f16(
            false, a1.v, false, b[t].v, (short)0, acc[1][t], false, false);
    }

    // D layout: VGPR r -> M = (lane>=16)*8 + r, N = lane&15
    const int scol = n0 + (lid & 15);
#pragma unroll
    for (int g = 0; g < 2; ++g) {
        const int srow = m0 + g * 16 + (half16 << 3);
#pragma unroll
        for (int t = 0; t < 4; ++t) {
            const int col = scol + t * 16;
            const float bv = bias[col];
#pragma unroll
            for (int r = 0; r < 8; ++r) {
                float x = acc[g][t][r] + bv;
                if (do_relu) x = x > 0.0f ? x : 0.0f;
                C[(size_t)(srow + r) * N + col] = (_Float16)x;
            }
        }
    }
}

// ------------------------------------------------ output head
// wave-per-row dot(h3[b,:256], Wo) + partial[b] + bo
__global__ __launch_bounds__(256) void final_out(
    const _Float16* __restrict__ h3, const float* __restrict__ Wo,
    const float* __restrict__ bo,    const float* __restrict__ partial,
    float* __restrict__ out)
{
    const int wv   = threadIdx.x >> 5;
    const int lane = threadIdx.x & 31;
    const int b    = blockIdx.x * 8 + wv;
    const _Float16* h = h3 + (size_t)b * H3;
    float s = 0.0f;
#pragma unroll
    for (int j = 0; j < 8; ++j) {
        int c = lane + j * 32;
        s += (float)h[c] * Wo[c];
    }
#pragma unroll
    for (int off = 16; off > 0; off >>= 1)
        s += __shfl_down(s, off, 32);
    if (lane == 0) out[b] = partial[b] + s + bo[0];
}

// ---------------------------------------------------------------- launch
extern "C" void kernel_launch(void* const* d_in, const int* in_sizes, int n_in,
                              void* d_out, int out_size, void* d_ws, size_t ws_size,
                              hipStream_t stream) {
    const float* x_num   = (const float*)d_in[0];
    const int*   x_cat   = (const int*)  d_in[1];
    const float* bias    = (const float*)d_in[2];
    const float* w_num   = (const float*)d_in[3];
    const float* lin_cat = (const float*)d_in[4];
    const float* v_num   = (const float*)d_in[5];
    const float* v_cat   = (const float*)d_in[6];
    const float* W1      = (const float*)d_in[7];
    const float* b1      = (const float*)d_in[8];
    const float* W2      = (const float*)d_in[9];
    const float* b2      = (const float*)d_in[10];
    const float* W3      = (const float*)d_in[11];
    const float* b3      = (const float*)d_in[12];
    const float* Wo      = (const float*)d_in[13];
    const float* bo      = (const float*)d_in[14];
    float* out = (float*)d_out;

    char* ws = (char*)d_ws;
    _Float16* deep  = (_Float16*)(ws + OFF_DEEP);
    _Float16* hA    = (_Float16*)(ws + OFF_HA);
    _Float16* W1t   = (_Float16*)(ws + OFF_W1T);
    _Float16* W2t   = (_Float16*)(ws + OFF_W2T);
    _Float16* W3t   = (_Float16*)(ws + OFF_W3T);
    float*    part  = (float*)   (ws + OFF_PART);
    _Float16* h1 = hA;      // [B,1024]
    _Float16* h2 = deep;    // alias: deep_in dead after layer 1
    _Float16* h3 = hA;      // alias: h1 dead after layer 2

    // weights -> f16 transposed/padded
    transpose_cvt<<<((size_t)H1 * KP1 + 255) / 256, 256, 0, stream>>>(W1, W1t, DEEP_IN, H1, KP1);
    transpose_cvt<<<((size_t)H2 * H1  + 255) / 256, 256, 0, stream>>>(W2, W2t, H1, H2, H1);
    transpose_cvt<<<((size_t)H3 * H2  + 255) / 256, 256, 0, stream>>>(W3, W3t, H2, H3, H2);

    // gather + FM first/second order + deep_in materialization
    build_features<<<BB, 64, 0, stream>>>(x_num, x_cat, bias, w_num, lin_cat,
                                          v_num, v_cat, deep, part);

    // MLP via WMMA (block tile 128x128)
    gemm_relu_f16<<<dim3(H1 / 128, BB / 128), 256, 0, stream>>>(deep, W1t, b1, h1, BB, H1, KP1, 1);
    gemm_relu_f16<<<dim3(H2 / 128, BB / 128), 256, 0, stream>>>(h1,   W2t, b2, h2, BB, H2, H1, 1);
    gemm_relu_f16<<<dim3(H3 / 128, BB / 128), 256, 0, stream>>>(h2,   W3t, b3, h3, BB, H3, H2, 1);

    // head
    final_out<<<BB / 8, 256, 0, stream>>>(h3, Wo, bo, part, out);
}
